// DIDAModule_19542101196953
// MI455X (gfx1250) — matrix operand order repaired and checked by reading, not compiled, not run on hardware
//
#include <hip/hip_runtime.h>
#include <hip/hip_bf16.h>

// ---------- problem dims ----------
#define NB   32
#define Cc   512
#define Hh   56
#define Ww   56
#define HW   3136      // 56*56
#define CMd  128
#define P2d  256

typedef __attribute__((ext_vector_type(16))) __bf16 v16bf;
typedef __attribute__((ext_vector_type(8)))  __bf16 v8bf;
typedef __attribute__((ext_vector_type(8)))  float  v8f;
typedef unsigned int v4u __attribute__((ext_vector_type(4)));
typedef int          v4i __attribute__((ext_vector_type(4)));
typedef int          v8i __attribute__((ext_vector_type(8)));

__device__ __forceinline__ __bf16 f2bf(float f) { return (__bf16)f; }
__device__ __forceinline__ float  bf2f(__bf16 h) { return (float)h; }

__device__ __forceinline__ v16bf combine16(v8bf lo, v8bf hi) {
    v16bf r;
#pragma unroll
    for (int i = 0; i < 8; ++i) { r[i] = lo[i]; r[8 + i] = hi[i]; }
    return r;
}

// ---------------- Tensor Data Mover (D# per ISA cdna5 §8) ----------------
#if __has_builtin(__builtin_amdgcn_tensor_load_to_lds) && \
    __has_builtin(__builtin_amdgcn_s_wait_tensorcnt)
#define HAVE_TDM 1
#else
#define HAVE_TDM 0
#endif

// g1 word0 flag sets: data_size=2B (1<<16); optional LDS pad: enable(1<<20),
// interval code 4 => every 32 DWORDs (4<<22), amount code 3 => 4 DWORDs (3<<25)
#define TDM_FLAGS_2B        (1u << 16)
#define TDM_FLAGS_2B_PAD32  ((1u << 16) | (1u << 20) | (4u << 22) | (3u << 25))

#if HAVE_TDM
// 2D tile load: rows x tile_w elements (2B each), global row stride = stride_elems.
// tensor dims == tile dims (no OOB use). Issue from ONE wave only (EXEC ignored).
__device__ __forceinline__ void tdm_load_2d(const void* gsrc, void* ldsDst,
                                            unsigned tile_w, unsigned rows,
                                            unsigned stride_elems, unsigned flags)
{
    unsigned lds_addr = (unsigned)(size_t)ldsDst;
    unsigned long long ga = (unsigned long long)(size_t)gsrc;
    v4u g0;
    g0[0] = 1u;                                   // count=1 (valid), user mode
    g0[1] = lds_addr;                             // [63:32] lds_addr
    g0[2] = (unsigned)(ga & 0xffffffffu);         // [95:64] global_addr lo
    g0[3] = (unsigned)((ga >> 32) & 0x01ffffffu)  // [120:96] global_addr hi
          | (2u << 30);                           // [127:126] type = 2 (image)
    v8i g1;
    g1[0] = (int)flags;                           // wgmask=0, data_size, pad cfg
    g1[1] = (int)(tile_w << 16);                  // [79:48] tensor_dim0 (lo16)
    g1[2] = (int)((tile_w >> 16) | (rows << 16)); // dim0 hi | tensor_dim1 lo16
    g1[3] = (int)((rows >> 16) | (tile_w << 16)); // dim1 hi | tile_dim0
    g1[4] = (int)rows;                            // tile_dim1 | tile_dim2=0
    g1[5] = (int)stride_elems;                    // tensor_dim0_stride lo32
    g1[6] = 0;                                    // stride hi | dim1_stride lo
    g1[7] = 0;
    v4i z = {0, 0, 0, 0};
#if defined(__clang_major__) && (__clang_major__ >= 23)
    v8i z8 = {0, 0, 0, 0, 0, 0, 0, 0};
    __builtin_amdgcn_tensor_load_to_lds(g0, g1, z, z, z8, 0);
#else
    __builtin_amdgcn_tensor_load_to_lds(g0, g1, z, z, 0);
#endif
}
#endif

// ============================================================
// Kernel 0: prep — bf16 copies of the two GEMM weight matrices
// ============================================================
__global__ __launch_bounds__(256)
void prep_kernel(const float* __restrict__ W_conv, const float* __restrict__ W_fuse,
                 __bf16* __restrict__ Wcb, __bf16* __restrict__ Wfb)
{
    int i = blockIdx.x * 256 + threadIdx.x;
    if (i < CMd * Cc)  Wcb[i] = f2bf(W_conv[i]);   // [128][512]
    if (i < P2d * CMd) Wfb[i] = f2bf(W_fuse[i]);   // [256][128]
}

// ============================================================
// Kernel 1: f = relu(x @ W_conv^T + b)  (bf16 out) + x partial sums
//   grid (14, 32), block 256 (8 waves). Tile: 224 px x 128 ch, K=512.
// ============================================================
#define T1PIX   224
#define KC      64
#define APITCH  72      // bf16 elems per As row (16B-aligned rows)
#define BPITCH  72      // == 64 data + 4-DWORD TDM pad per 32-DWORD row

__global__ __launch_bounds__(256)
void fconv_kernel(const float* __restrict__ x, const __bf16* __restrict__ Wcb,
                  const float* __restrict__ b_conv, __bf16* __restrict__ fout,
                  float* __restrict__ part /* [14][32][512][4] */)
{
    __shared__ __bf16 As[T1PIX * APITCH];   // 32,256 B : [pixel][k]
    __shared__ __bf16 Bs[CMd  * BPITCH];    // 18,432 B : [outch][k]

    const int n    = blockIdx.y;
    const int bx   = blockIdx.x;
    const int pix0 = bx * T1PIX;
    const int t    = threadIdx.x;
    const int lane = t & 31;
    const int wv   = t >> 5;        // wave id 0..7 == n-tile
    const int col  = lane & 15;
    const int hi   = lane >> 4;

    v8f acc[14];
#pragma unroll
    for (int i = 0; i < 14; ++i) acc[i] = {};

    const int lk = t >> 2;   // channel within chunk 0..63
    const int lq = t & 3;    // pixel quarter

    for (int kc = 0; kc < 8; ++kc) {
        __syncthreads();
        // ---- B chunk: Wcb[0..127][kc*64 ..+63] -> Bs (pitch 72 via TDM pad)
#if HAVE_TDM
        if (t < 32) {   // one wave issues the DMA (TDM ignores EXEC)
            tdm_load_2d(Wcb + kc * KC, Bs, KC, CMd, Cc, TDM_FLAGS_2B_PAD32);
        }
#else
        {
            const int o = t >> 1, hb = t & 1;
            const __bf16* wp = Wcb + (size_t)o * Cc + kc * KC + hb * 32;
            uint4* dst = (uint4*)(&Bs[o * BPITCH + hb * 32]);
            dst[0] = *(const uint4*)(wp);
            dst[1] = *(const uint4*)(wp + 8);
            dst[2] = *(const uint4*)(wp + 16);
            dst[3] = *(const uint4*)(wp + 24);
        }
#endif
        // ---- A chunk: x[n][kc*64+lk][pix0 + lq*56 .. +55] -> As[m][k], + psum
        {
            const float* xp = x + ((size_t)(n * Cc + kc * KC + lk)) * HW + pix0 + lq * 56;
            if (kc < 7) __builtin_prefetch(xp + (size_t)KC * HW, 0, 1);
            float psum = 0.f;
#pragma unroll
            for (int j = 0; j < 14; ++j) {
                float4 v = *(const float4*)(xp + j * 4);
                psum += v.x + v.y + v.z + v.w;
                int m = lq * 56 + j * 4;
                As[(m + 0) * APITCH + lk] = f2bf(v.x);
                As[(m + 1) * APITCH + lk] = f2bf(v.y);
                As[(m + 2) * APITCH + lk] = f2bf(v.z);
                As[(m + 3) * APITCH + lk] = f2bf(v.w);
            }
            part[(((size_t)bx * NB + n) * Cc + kc * KC + lk) * 4 + lq] = psum;
        }
#if HAVE_TDM
        if (t < 32) __builtin_amdgcn_s_wait_tensorcnt(0);
#endif
        __syncthreads();
        // ---- compute: wave wv owns output channels wv*16..+15, all 14 m-tiles
#pragma unroll
        for (int ks = 0; ks < 2; ++ks) {
            const __bf16* bp = &Bs[(wv * 16 + col) * BPITCH + ks * 32 + hi * 8];
            v16bf bfrag = combine16(*(const v8bf*)bp, *(const v8bf*)(bp + 16));
#pragma unroll
            for (int mt = 0; mt < 14; ++mt) {
                const __bf16* ap = &As[(mt * 16 + col) * APITCH + ks * 32 + hi * 8];
                v16bf afrag = combine16(*(const v8bf*)ap, *(const v8bf*)(ap + 16));
                acc[mt] = __builtin_amdgcn_wmma_f32_16x16x32_bf16(
                    false, afrag, false, bfrag, (short)0, acc[mt], false, false);
            }
        }
    }
    __syncthreads();   // As about to be reused as staging

    // ---- epilogue: bias + relu, transpose via LDS, b128 bf16 stores
    const float bias = b_conv[wv * 16 + col];
    float* stg = (float*)&As[0] + wv * (16 * 20);   // 1280 B per wave
    const int sch = lane >> 1, shalf = lane & 1;
#pragma unroll
    for (int mt = 0; mt < 14; ++mt) {
#pragma unroll
        for (int r = 0; r < 8; ++r) {
            float v = acc[mt][r] + bias;
            stg[col * 20 + r + 8 * hi] = v > 0.f ? v : 0.f;   // [ch][m]
        }
        v8bf pk;
#pragma unroll
        for (int j = 0; j < 8; ++j) pk[j] = f2bf(stg[sch * 20 + shalf * 8 + j]);
        __bf16* fp = fout + ((size_t)(n * CMd + wv * 16 + sch)) * HW
                          + pix0 + mt * 16 + shalf * 8;
        *(v8bf*)fp = pk;
    }
}

// ============================================================
// Kernel 2: g = relu(mean(x) @ W^T + b) -> dynamic kernels
//   grid 32, block 128 (thread = cm)
// ============================================================
__global__ __launch_bounds__(128)
void gkern_kernel(const float* __restrict__ part, const float* __restrict__ W_conv,
                  const float* __restrict__ b_conv,
                  const float* __restrict__ wk,  const float* __restrict__ bk,
                  const float* __restrict__ wck, const float* __restrict__ bck,
                  const float* __restrict__ wk2, const float* __restrict__ bk2,
                  const float* __restrict__ wck2,const float* __restrict__ bck2,
                  float* __restrict__ kern1, float* __restrict__ kern2)
{
    __shared__ float xs_s[Cc];
    const int n = blockIdx.x, cm = threadIdx.x;
    for (int c = cm; c < Cc; c += 128) {
        float s = 0.f;
        for (int b = 0; b < 14; ++b) {
            const float* pp = part + (((size_t)b * NB + n) * Cc + c) * 4;
            s += pp[0] + pp[1] + pp[2] + pp[3];
        }
        xs_s[c] = s;
    }
    __syncthreads();
    const float* wr = W_conv + (size_t)cm * Cc;
    float acc = 0.f;
    for (int c = 0; c < Cc; ++c) acc += xs_s[c] * wr[c];
    float g = acc * (1.f / 3136.f) + b_conv[cm];
    g = g > 0.f ? g : 0.f;
    const float a1 = wck[0], c1 = bck[0];
    float* k1 = kern1 + ((size_t)n * CMd + cm) * 25;
#pragma unroll
    for (int i = 0; i < 25; ++i) k1[i] = (g * wk[i] + bk[i]) * a1 + c1;
    const float a2 = wck2[0], c2 = bck2[0];
    float* k2 = kern2 + ((size_t)n * CMd + cm) * 9;
#pragma unroll
    for (int i = 0; i < 9; ++i) k2[i] = (g * wk2[i] + bk2[i]) * a2 + c2;
}

// ============================================================
// Kernel 3: depthwise (both branches) + fuse GEMM + concat store
//   grid (28, 32), block 256, dynamic LDS 146,176 B
// ============================================================
#define TILE3   112
#define HOFF    114     // 2*56 + 2 halo
#define RPITCH  344     // 340 region px padded to mult-of-8
#define AOPITCH 136

#define SM_STG   0
#define SM_K1    10240
#define SM_K2    23040
#define SM_AO    27648
#define SM_FS    58112
#define SM_TOTAL 146176

__global__ __launch_bounds__(256)
void dida_tail_kernel(const __bf16* __restrict__ fin, const __bf16* __restrict__ Wfb,
                      const float* __restrict__ kern1, const float* __restrict__ kern2,
                      const float* __restrict__ b_fuse, float* __restrict__ out)
{
    extern __shared__ char smem_raw[];
    float*  stg = (float*)(smem_raw + SM_STG);    // 8 waves * 16x20 f32
    float*  k1s = (float*)(smem_raw + SM_K1);     // 128*25
    float*  k2s = (float*)(smem_raw + SM_K2);     // 128*9
    __bf16* Ao  = (__bf16*)(smem_raw + SM_AO);    // [112][136]
    __bf16* fs  = (__bf16*)(smem_raw + SM_FS);    // [128][344]

    const int n    = blockIdx.y;
    const int pix0 = blockIdx.x * TILE3;
    const int t    = threadIdx.x;
    const int lane = t & 31;
    const int wv   = t >> 5;
    const int col  = lane & 15;
    const int hi   = lane >> 4;

    // dynamic kernel taps for this n
    for (int i = t; i < CMd * 25; i += 256) k1s[i] = kern1[(size_t)n * CMd * 25 + i];
    for (int i = t; i < CMd * 9;  i += 256) k2s[i] = kern2[(size_t)n * CMd * 9  + i];

    // f halo region: 128 ch x 344 px.  Interior tiles: one 2D TDM descriptor.
    {
        const int g0 = pix0 - HOFF;
        bool interior = (g0 >= 0) && (g0 + RPITCH <= HW);
#if HAVE_TDM
        if (interior) {
            if (t < 32) {
                tdm_load_2d(fin + (size_t)n * CMd * HW + g0, fs,
                            RPITCH, CMd, HW, TDM_FLAGS_2B);
                __builtin_amdgcn_s_wait_tensorcnt(0);
            }
        } else
#endif
        {
            for (int i = t; i < CMd * (RPITCH / 2); i += 256) {
                int ch = i / (RPITCH / 2);
                int j  = i - ch * (RPITCH / 2);
                int p0 = g0 + j * 2;
                const __bf16* fp = fin + ((size_t)(n * CMd + ch)) * HW;
                if (p0 >= 0 && p0 + 1 < HW) {
                    *(unsigned*)(fs + ch * RPITCH + j * 2) = *(const unsigned*)(fp + p0);
                } else {
                    fs[ch * RPITCH + j * 2]     = (p0     >= 0 && p0     < HW) ? fp[p0]     : f2bf(0.f);
                    fs[ch * RPITCH + j * 2 + 1] = (p0 + 1 >= 0 && p0 + 1 < HW) ? fp[p0 + 1] : f2bf(0.f);
                }
            }
        }
    }
    __syncthreads();

    for (int branch = 0; branch < 2; ++branch) {
        // ---- depthwise: o(ch, m) -> Ao[m][ch] (bf16, GEMM-A layout)
        {
            const int ch   = t & 127;
            const int half = t >> 7;
            const float*  kp   = (branch == 0) ? (k1s + ch * 25) : (k2s + ch * 9);
            const __bf16* frow = fs + ch * RPITCH;
            const int KK  = (branch == 0) ? 5 : 3;
            const int dil = (branch == 0) ? 1 : 2;
            for (int mi = 0; mi < 56; ++mi) {
                int m  = half * 56 + mi;
                int gp = pix0 + m;
                int h  = gp / Ww;
                int w  = gp - h * Ww;
                float acc = 0.f;
                for (int ky = 0; ky < KK; ++ky) {
                    int dh = (ky - (KK >> 1)) * dil;
                    int hh = h + dh;
                    if (hh < 0 || hh >= Hh) continue;
                    for (int kx = 0; kx < KK; ++kx) {
                        int dw = (kx - (KK >> 1)) * dil;
                        int w2 = w + dw;
                        if (w2 < 0 || w2 >= Ww) continue;
                        acc += kp[ky * KK + kx] * bf2f(frow[m + HOFF + dh * Ww + dw]);
                    }
                }
                Ao[m * AOPITCH + ch] = f2bf(acc);
            }
        }
        __syncthreads();

        // ---- fuse GEMM: M=112(7 tiles), N=256(16 tiles), K=128(4 steps)
        const int ob = branch * P2d;    // concat offset
#pragma unroll
        for (int nn = 0; nn < 2; ++nn) {
            const int nt  = wv * 2 + nn;
            const int chg = nt * 16 + col;
            const float bias = b_fuse[chg];
            v16bf bfr[4];
#pragma unroll
            for (int ks = 0; ks < 4; ++ks)   // 32B contiguous, L2-resident
                bfr[ks] = *(const v16bf*)(Wfb + (size_t)chg * CMd + ks * 32 + hi * 16);
#pragma unroll
            for (int mt = 0; mt < 7; ++mt) {
                v8f acc;
#pragma unroll
                for (int r = 0; r < 8; ++r) acc[r] = bias;
#pragma unroll
                for (int ks = 0; ks < 4; ++ks) {
                    const __bf16* ap = Ao + (mt * 16 + col) * AOPITCH + ks * 32 + hi * 16;
                    v16bf afrag = combine16(*(const v8bf*)ap, *(const v8bf*)(ap + 8));
                    acc = __builtin_amdgcn_wmma_f32_16x16x32_bf16(
                        false, afrag, false, bfr[ks], (short)0, acc, false, false);
                }
                // transpose D tile via per-wave LDS, store b128 runs
                float* s = stg + wv * (16 * 20);
#pragma unroll
                for (int r = 0; r < 8; ++r) s[col * 20 + r + 8 * hi] = acc[r];
                const int sch = lane >> 1, shalf = lane & 1;
                float4 lo4 = *(const float4*)(s + sch * 20 + shalf * 8);
                float4 hi4 = *(const float4*)(s + sch * 20 + shalf * 8 + 4);
                float* op = out + ((size_t)(n * 512 + ob + nt * 16 + sch)) * HW
                                + pix0 + mt * 16 + shalf * 8;
                *(float4*)op       = lo4;
                *((float4*)op + 1) = hi4;
            }
        }
        __syncthreads();   // Ao reused by next branch
    }
}

// ============================================================
// launcher
// ============================================================
extern "C" void kernel_launch(void* const* d_in, const int* in_sizes, int n_in,
                              void* d_out, int out_size, void* d_ws, size_t ws_size,
                              hipStream_t stream)
{
    (void)in_sizes; (void)n_in; (void)out_size; (void)ws_size;
    const float* x      = (const float*)d_in[0];
    const float* W_conv = (const float*)d_in[1];
    const float* b_conv = (const float*)d_in[2];
    const float* wk     = (const float*)d_in[3];
    const float* bk     = (const float*)d_in[4];
    const float* wck    = (const float*)d_in[5];
    const float* bck    = (const float*)d_in[6];
    const float* wk2    = (const float*)d_in[7];
    const float* bk2    = (const float*)d_in[8];
    const float* wck2   = (const float*)d_in[9];
    const float* bck2   = (const float*)d_in[10];
    const float* W_fuse = (const float*)d_in[11];
    const float* b_fuse = (const float*)d_in[12];
    float* out = (float*)d_out;

    char* ws = (char*)d_ws;
    __bf16* fbuf  = (__bf16*)(ws);                 // 32*128*3136*2 = 25,690,112
    __bf16* Wcb   = (__bf16*)(ws + 25690112);      // 131,072
    __bf16* Wfb   = (__bf16*)(ws + 25821184);      // 65,536
    float*  part  = (float*) (ws + 25886720);      // 14*32*512*4*4 = 3,670,016
    float*  kern1 = (float*) (ws + 29556736);      // 409,600
    float*  kern2 = (float*) (ws + 29966336);      // 147,456  (end ~30.1 MB)

    prep_kernel<<<256, 256, 0, stream>>>(W_conv, W_fuse, Wcb, Wfb);
    fconv_kernel<<<dim3(14, 32), 256, 0, stream>>>(x, Wcb, b_conv, fbuf, part);
    gkern_kernel<<<32, 128, 0, stream>>>(part, W_conv, b_conv, wk, bk, wck, bck,
                                         wk2, bk2, wck2, bck2, kern1, kern2);
    dida_tail_kernel<<<dim3(28, 32), 256, SM_TOTAL, stream>>>(fbuf, Wfb, kern1, kern2,
                                                              b_fuse, out);
}
